// Attention_2705829397050
// MI455X (gfx1250) — matrix operand order repaired and checked by reading, not compiled, not run on hardware
//
#include <hip/hip_runtime.h>

// ---------------------------------------------------------------------------
// MHA forward for MI455X (gfx1250, wave32, WMMA 16x16x32 bf16).
// Compute-bound (~120 GFLOP vs ~125 MB traffic) -> all matmuls via
// v_wmma_f32_16x16x32_bf16 (f32 accum). Attention stages K/V/mask tiles into
// LDS with double-buffered global_load_async_to_lds_b128 (ASYNCcnt path),
// shared by all 8 waves of the block.
// ---------------------------------------------------------------------------

#define DMODEL 1024
#define NHEAD  16
#define DK     64
#define SEQ    2048
#define BATCH  4
#define BH     (BATCH * NHEAD)

typedef __attribute__((ext_vector_type(16))) __bf16 v16bf;
typedef __attribute__((ext_vector_type(8)))  __bf16 bf16x8;
typedef __attribute__((ext_vector_type(8)))  float  v8f;
typedef __attribute__((ext_vector_type(4)))  float  f32x4;

union Frag16 { v16bf v; bf16x8 h[2]; __bf16 e[16]; };
union Acc8   { v8f   v; float  f[8]; };

static __device__ inline v8f wmma_bf16(v16bf a, v16bf b, v8f c) {
  // D = A(16x32 bf16) x B(32x16 bf16) + C(16x16 f32)
  return __builtin_amdgcn_wmma_f32_16x16x32_bf16(
      /*neg_a=*/false, a, /*neg_b=*/false, b,
      /*c_mod=*/(short)0, c, /*reuse_a=*/false, /*reuse_b=*/false);
}

// Async copy 16 bytes global -> LDS (ASYNCcnt-tracked, per ISA 10.x).
static __device__ inline void async_b128_to_lds(void* lds_ptr, const void* gptr) {
  unsigned int       l = (unsigned int)(unsigned long long)(uintptr_t)lds_ptr;
  unsigned long long g = (unsigned long long)(uintptr_t)gptr;
  asm volatile("global_load_async_to_lds_b128 %0, %1, off"
               :: "v"(l), "v"(g) : "memory");
}
static __device__ inline void wait_async0() {
  asm volatile("s_wait_asynccnt 0x0" ::: "memory");
}

// ---------------------------------------------------------------------------
// Kernel 0: W (f32, [K][N] row-major) -> Wt (bf16, [N][K]) transpose-convert.
// Makes the WMMA B-fragment (two 8-K runs, 16 apart, at fixed N) two b128 loads.
// ---------------------------------------------------------------------------
__global__ void __launch_bounds__(256)
wconv_kernel(const float* __restrict__ W, __bf16* __restrict__ Wt) {
  int idx = blockIdx.x * 256 + threadIdx.x;   // 0 .. 1024*1024-1
  int n = idx >> 10;
  int k = idx & 1023;
  Wt[idx] = (__bf16)W[(size_t)k * DMODEL + n];
}

// ---------------------------------------------------------------------------
// Kernel 1: projection  dst = X @ W  (one wave per 16-row x 64-col tile).
// X: f32 [B*S][1024]. Wt: bf16 [N][K]. colGroup == head (64 cols per head).
// transposed==0: dst[bh][s][d] bf16 (Q/K).  transposed==1: dst[bh][d][s] (V).
// ---------------------------------------------------------------------------
__global__ void __launch_bounds__(256)
proj_kernel(const float* __restrict__ X, const __bf16* __restrict__ Wt,
            __bf16* __restrict__ dst, float scale, int transposed) {
  const int lane   = threadIdx.x & 31;
  const int waveId = threadIdx.x >> 5;
  const int w      = blockIdx.x * 8 + waveId;  // 8192 wave-tiles
  const int rowTile = w >> 4;                  // 0..511 (16 rows each)
  const int h       = w & 15;                  // head == column group
  const int r = lane & 15;                     // A row / B column / C column
  const int g = lane >> 4;                     // half-wave select

  const int row = rowTile * 16 + r;            // 0..8191
  const float* xr = X + (size_t)row * DMODEL;

  Acc8 acc[4];
#pragma unroll
  for (int t = 0; t < 4; ++t)
#pragma unroll
    for (int i = 0; i < 8; ++i) acc[t].f[i] = 0.0f;

  for (int k0 = 0; k0 < DMODEL; k0 += 32) {
    // A fragment: lane holds K = {8g..8g+7} then {16+8g..23+8g} (ISA layout).
    f32x4 x0 = *(const f32x4*)(xr + k0 + 8 * g);
    f32x4 x1 = *(const f32x4*)(xr + k0 + 8 * g + 4);
    f32x4 x2 = *(const f32x4*)(xr + k0 + 16 + 8 * g);
    f32x4 x3 = *(const f32x4*)(xr + k0 + 16 + 8 * g + 4);
    Frag16 a;
#pragma unroll
    for (int i = 0; i < 4; ++i) {
      a.e[i]      = (__bf16)x0[i];
      a.e[4 + i]  = (__bf16)x1[i];
      a.e[8 + i]  = (__bf16)x2[i];
      a.e[12 + i] = (__bf16)x3[i];
    }
#pragma unroll
    for (int t = 0; t < 4; ++t) {
      // B fragment: column n = 16t+r of this head, K = k0+8g..+7 then k0+16+8g..+7.
      const __bf16* wr = Wt + (size_t)(h * 64 + 16 * t + r) * DMODEL + k0 + 8 * g;
      Frag16 bfr;
      bfr.h[0] = *(const bf16x8*)wr;
      bfr.h[1] = *(const bf16x8*)(wr + 16);
      acc[t].v = wmma_bf16(a.v, bfr.v, acc[t].v);
    }
  }

  const int sBase = (rowTile * 16) & (SEQ - 1);
  const int bIdx  = (rowTile * 16) >> 11;
  const int bh    = bIdx * NHEAD + h;

  if (!transposed) {
    // C layout: lane (n=r, g), VGPR v -> row M = 8g+v, col n. Row-major store.
#pragma unroll
    for (int t = 0; t < 4; ++t)
#pragma unroll
      for (int vv = 0; vv < 8; ++vv) {
        int s = sBase + 8 * g + vv;
        dst[((size_t)bh * SEQ + s) * DK + 16 * t + r] =
            (__bf16)(acc[t].f[vv] * scale);
      }
  } else {
    // Transposed store: fixed d = 16t+r, s contiguous -> one b128 per subtile.
#pragma unroll
    for (int t = 0; t < 4; ++t) {
      bf16x8 pk;
#pragma unroll
      for (int vv = 0; vv < 8; ++vv) pk[vv] = (__bf16)(acc[t].f[vv] * scale);
      *(bf16x8*)(dst + ((size_t)bh * DK + 16 * t + r) * SEQ + sBase + 8 * g) = pk;
    }
  }
}

// ---------------------------------------------------------------------------
// Kernel 2: flash attention. One wave owns 16 Q rows; 32-col KV tiles.
// Qh/Kh: bf16 [bh][s][64] (Q pre-scaled by 0.125). Vt: bf16 [bh][d][s].
// K/V/mask tiles double-buffered in LDS via async-to-LDS copies.
// ---------------------------------------------------------------------------
__global__ void __launch_bounds__(256)
attn_kernel(const __bf16* __restrict__ Qh, const __bf16* __restrict__ Kh,
            const __bf16* __restrict__ Vt, const unsigned char* __restrict__ mask,
            float* __restrict__ out) {
  __shared__ __bf16        ldsK[2][32 * 64];       // [buf][kv_row][d]   4KB each
  __shared__ __bf16        ldsV[2][64 * 32];       // [buf][d][kv_s]     4KB each
  __shared__ unsigned char ldsM[2][8 * 16 * 32];   // [buf][wave][qr][c] 4KB each
  __shared__ __bf16        ldsP[8 * 16 * 32];      // per-wave P relayout 8KB

  const int tid  = threadIdx.x;
  const int lane = tid & 31;
  const int wave = tid >> 5;
  const int bh   = blockIdx.x >> 4;
  const int qBlk = blockIdx.x & 15;
  const int q0   = qBlk * 128 + wave * 16;
  const int b    = bh >> 4;
  const int r    = lane & 15;
  const int g    = lane >> 4;
  const int n    = r;

  // Global bases for the block's shared KV / mask tiles.
  const __bf16* Kbase = Kh + (size_t)bh * SEQ * DK;         // + (kv0+row)*64
  const __bf16* Vbase = Vt + (size_t)bh * DK * SEQ;         // + d*2048 + kv0
  const unsigned char* Mbase = mask + ((size_t)b * SEQ + qBlk * 128) * SEQ;

  // Cooperative async stage of one 32-column KV tile + mask tile (12KB/block):
  //   K: 32 rows x 128B  -> thread t: row=t>>3, 16B chunk c=t&7
  //   V: 64 rows x  64B  -> thread t: d  =t>>2, 16B chunk c=t&3
  //   M: 8 waves x 16 q-rows x 32B -> thread t: 16B chunk t
  auto issue_tile = [&](int kv0, int buf) {
    {
      int row = tid >> 3, c = tid & 7;
      async_b128_to_lds(&ldsK[buf][row * DK + c * 8],
                        Kbase + (size_t)(kv0 + row) * DK + c * 8);
    }
    {
      int d = tid >> 2, c = tid & 3;
      async_b128_to_lds(&ldsV[buf][d * 32 + c * 8],
                        Vbase + (size_t)d * SEQ + kv0 + c * 8);
    }
    {
      int w2 = tid >> 5, rem = tid & 31, qr = rem >> 1, half = rem & 1;
      async_b128_to_lds(&ldsM[buf][tid * 16],
                        Mbase + (size_t)(w2 * 16 + qr) * SEQ + kv0 + half * 16);
    }
  };

  // Q A-fragments, resident for the whole kernel (d 0..31 and 32..63).
  const __bf16* qrow = Qh + ((size_t)bh * SEQ + q0 + r) * DK;
  Frag16 qa0, qa1;
  qa0.h[0] = *(const bf16x8*)(qrow + 16 * g);
  qa0.h[1] = *(const bf16x8*)(qrow + 16 * g + 8);
  qa1.h[0] = *(const bf16x8*)(qrow + 32 + 16 * g);
  qa1.h[1] = *(const bf16x8*)(qrow + 40 + 16 * g);

  float mrun[8], lrun[8];
  Acc8 o[4];
#pragma unroll
  for (int vv = 0; vv < 8; ++vv) { mrun[vv] = -3.0e38f; lrun[vv] = 0.0f; }
#pragma unroll
  for (int t = 0; t < 4; ++t)
#pragma unroll
    for (int vv = 0; vv < 8; ++vv) o[t].f[vv] = 0.0f;

  __bf16* pls       = ldsP + wave * 512;
  const __bf16* plr = ldsP + wave * 512 + r * 32;

  // Prologue: stage tile 0, wait for async writes, make visible to all waves.
  issue_tile(0, 0);
  wait_async0();
  __syncthreads();

  for (int kv0 = 0; kv0 < SEQ; kv0 += 32) {
    const int buf = (kv0 >> 5) & 1;
    if (kv0 + 32 < SEQ) issue_tile(kv0 + 32, buf ^ 1);  // overlap next tile

    // ---- S = Q . K^T for 32 kv columns (two 16x16 C tiles), K from LDS ----
    Acc8 s0, s1;
#pragma unroll
    for (int vv = 0; vv < 8; ++vv) { s0.f[vv] = 0.0f; s1.f[vv] = 0.0f; }
    {
      const __bf16* kr0 = &ldsK[buf][n * DK];          // kv col n
      const __bf16* kr1 = &ldsK[buf][(16 + n) * DK];   // kv col 16+n
      Frag16 kb;
      kb.h[0] = *(const bf16x8*)(kr0 + 16 * g);
      kb.h[1] = *(const bf16x8*)(kr0 + 16 * g + 8);
      s0.v = wmma_bf16(qa0.v, kb.v, s0.v);
      kb.h[0] = *(const bf16x8*)(kr0 + 32 + 16 * g);
      kb.h[1] = *(const bf16x8*)(kr0 + 40 + 16 * g);
      s0.v = wmma_bf16(qa1.v, kb.v, s0.v);
      kb.h[0] = *(const bf16x8*)(kr1 + 16 * g);
      kb.h[1] = *(const bf16x8*)(kr1 + 16 * g + 8);
      s1.v = wmma_bf16(qa0.v, kb.v, s1.v);
      kb.h[0] = *(const bf16x8*)(kr1 + 32 + 16 * g);
      kb.h[1] = *(const bf16x8*)(kr1 + 40 + 16 * g);
      s1.v = wmma_bf16(qa1.v, kb.v, s1.v);
    }

    // ---- mask (from LDS) + online softmax (16-lane xor reductions) ----
    const unsigned char* mw = &ldsM[buf][wave * 512];
#pragma unroll
    for (int vv = 0; vv < 8; ++vv) {
      float sv0 = mw[(8 * g + vv) * 32 + n]      ? s0.f[vv] : -1.0e9f;
      float sv1 = mw[(8 * g + vv) * 32 + 16 + n] ? s1.f[vv] : -1.0e9f;
      float mx = fmaxf(sv0, sv1);
#pragma unroll
      for (int off = 1; off < 16; off <<= 1)
        mx = fmaxf(mx, __shfl_xor(mx, off, 32));
      float mnew  = fmaxf(mrun[vv], mx);
      float alpha = __expf(mrun[vv] - mnew);
      float p0 = __expf(sv0 - mnew);
      float p1 = __expf(sv1 - mnew);
      float rs = p0 + p1;
#pragma unroll
      for (int off = 1; off < 16; off <<= 1)
        rs += __shfl_xor(rs, off, 32);
      lrun[vv] = lrun[vv] * alpha + rs;
      mrun[vv] = mnew;
#pragma unroll
      for (int t = 0; t < 4; ++t) o[t].f[vv] *= alpha;
      // stash P in LDS (row-major 16x32) to relayout C -> A fragment
      pls[(8 * g + vv) * 32 + n]      = (__bf16)p0;
      pls[(8 * g + vv) * 32 + 16 + n] = (__bf16)p1;
    }

    // ---- O += P . V ----  (A = P 16x32 from LDS, B = V tile from LDS)
    Frag16 pa;
    pa.h[0] = *(const bf16x8*)(plr + 16 * g);
    pa.h[1] = *(const bf16x8*)(plr + 16 * g + 8);
#pragma unroll
    for (int t = 0; t < 4; ++t) {
      const __bf16* vr = &ldsV[buf][(16 * t + n) * 32 + 16 * g];
      Frag16 vb;
      vb.h[0] = *(const bf16x8*)vr;
      vb.h[1] = *(const bf16x8*)(vr + 8);
      o[t].v = wmma_bf16(pa.v, vb.v, o[t].v);
    }

    // Next tile's async writes complete + all waves done with this buffer.
    wait_async0();
    __syncthreads();
  }

  // ---- normalize by row sums and store f32 [B][H][S][64] ----
#pragma unroll
  for (int vv = 0; vv < 8; ++vv) {
    float invl = 1.0f / lrun[vv];
    float* orow = out + ((size_t)bh * SEQ + q0 + 8 * g + vv) * DK;
#pragma unroll
    for (int t = 0; t < 4; ++t)
      orow[16 * t + n] = o[t].f[vv] * invl;
  }
}

// ---------------------------------------------------------------------------
extern "C" void kernel_launch(void* const* d_in, const int* in_sizes, int n_in,
                              void* d_out, int out_size, void* d_ws, size_t ws_size,
                              hipStream_t stream) {
  const float* q  = (const float*)d_in[0];
  const float* k  = (const float*)d_in[1];
  const float* v  = (const float*)d_in[2];
  const unsigned char* mask = (const unsigned char*)d_in[3];
  const float* Wq = (const float*)d_in[4];
  const float* Wk = (const float*)d_in[5];
  const float* Wv = (const float*)d_in[6];
  float* out = (float*)d_out;

  // Workspace layout (bf16): 3x Wt (2 MB each) + Qh/Kh/Vt (16.8 MB each) ~ 57 MB
  __bf16* p   = (__bf16*)d_ws;
  __bf16* WtQ = p; p += (size_t)DMODEL * DMODEL;
  __bf16* WtK = p; p += (size_t)DMODEL * DMODEL;
  __bf16* WtV = p; p += (size_t)DMODEL * DMODEL;
  __bf16* Qh  = p; p += (size_t)BH * SEQ * DK;
  __bf16* Kh  = p; p += (size_t)BH * SEQ * DK;
  __bf16* Vt  = p; p += (size_t)BH * SEQ * DK;

  wconv_kernel<<<(DMODEL * DMODEL) / 256, 256, 0, stream>>>(Wq, WtQ);
  wconv_kernel<<<(DMODEL * DMODEL) / 256, 256, 0, stream>>>(Wk, WtK);
  wconv_kernel<<<(DMODEL * DMODEL) / 256, 256, 0, stream>>>(Wv, WtV);

  // 8192 wave-tiles / 8 waves per block = 1024 blocks per projection.
  proj_kernel<<<1024, 256, 0, stream>>>(q, WtQ, Qh, 0.125f, 0);  // Q / sqrt(dk)
  proj_kernel<<<1024, 256, 0, stream>>>(k, WtK, Kh, 1.0f, 0);
  proj_kernel<<<1024, 256, 0, stream>>>(v, WtV, Vt, 1.0f, 1);    // V transposed

  // B*H slices x 16 q-blocks of 128 rows, 8 waves x 16 rows per block.
  attn_kernel<<<BH * 16, 256, 0, stream>>>(Qh, Kh, Vt, mask, out);
}
